// NeuralCDE_21638045237319
// MI455X (gfx1250) — matrix operand order repaired and checked by reading, not compiled, not run on hardware
//
#include <hip/hip_runtime.h>

// ---------------- dims ----------------
#define BB 512
#define TT 512
#define DD 8
#define HH 128
#define WW 256
#define MR 32           // batch rows per workgroup (2 WMMA M-tiles)
#define DTF 0.1f

typedef _Float16 v16h __attribute__((ext_vector_type(16)));
typedef float    v8f  __attribute__((ext_vector_type(8)));

// Global-address-space fragment load -> global_load_b128 (not flat).
typedef __attribute__((address_space(1))) const _Float16 gas_chalf;
__device__ inline v16h load_bfrag(const _Float16* p) {
    const gas_chalf* gp = (const gas_chalf*)p;
    v16h r;
    #pragma unroll
    for (int i = 0; i < 16; ++i) r[i] = gp[i];
    return r;
}

// Branch-free activations (v_exp_f32 + v_rcp_f32 TRANS ops co-execute with WMMA).
__device__ inline float lipswish_f(float v) {
    float e = __expf(-v);
    return 0.909f * v * __builtin_amdgcn_rcpf(1.f + e);
}
__device__ inline float tanh_f(float v) {
    float e = __expf(2.f * v);          // inf -> +1 ; 0 -> -1 (saturates correctly)
    return 1.f - 2.f * __builtin_amdgcn_rcpf(e + 1.f);
}

// =====================================================================
// Prep kernel 1: dX[b,k,:] = X(t_{k+1}) - X(t_k) with float-accumulated t
// =====================================================================
__device__ inline void interp8(const float* tb, const float* yb, float t, float* x) {
    int lo = 0, hi = TT;
    while (lo < hi) { int mid = (lo + hi) >> 1; if (tb[mid] <= t) lo = mid + 1; else hi = mid; }
    int idx = lo - 1;
    if (idx < 0) idx = 0;
    if (idx > TT - 2) idx = TT - 2;
    float frac = (t - tb[idx]) / (tb[idx + 1] - tb[idx]);
    #pragma unroll
    for (int d = 0; d < DD; ++d) {
        float a = yb[(size_t)idx * DD + d];
        float b = yb[(size_t)(idx + 1) * DD + d];
        x[d] = a + frac * (b - a);
    }
}

__global__ void dx_kernel(const float* __restrict__ ts, const float* __restrict__ ys,
                          float* __restrict__ dX) {
    int b = blockIdx.x * blockDim.x + threadIdx.x;
    if (b >= BB) return;
    const float* tb = ts + (size_t)b * TT;
    const float* yb = ys + (size_t)b * TT * DD;
    float t = tb[0];
    float x0[DD], x1[DD];
    interp8(tb, yb, t, x0);
    for (int k = 0; k < TT; ++k) {
        float t1 = t + DTF;
        interp8(tb, yb, t1, x1);
        #pragma unroll
        for (int d = 0; d < DD; ++d) {
            dX[((size_t)b * TT + k) * DD + d] = x1[d] - x0[d];
            x0[d] = x1[d];
        }
        t = t1;
    }
}

// =====================================================================
// Prep: init MLP (plain f32; tiny cost)
// =====================================================================
__global__ void pack_init_kernel(const float* __restrict__ ts, const float* __restrict__ ys,
                                 float* __restrict__ out) {
    int i = blockIdx.x * blockDim.x + threadIdx.x;
    if (i >= BB * 9) return;
    int b = i / 9, j = i % 9;
    out[i] = (j == 0) ? ts[(size_t)b * TT] : ys[(size_t)b * TT * DD + (j - 1)];
}

__global__ void dense_f32_kernel(const float* __restrict__ X, int ldx,
                                 const float* __restrict__ Wm, const float* __restrict__ bias,
                                 float* __restrict__ Y, int K, int N, int act) {
    int i = blockIdx.x * blockDim.x + threadIdx.x;
    if (i >= BB * N) return;
    int b = i / N, n = i % N;
    const float* x = X + (size_t)b * ldx;
    const float* w = Wm + (size_t)n * K;
    float acc = bias[n];
    for (int kk = 0; kk < K; ++kk) acc += w[kk] * x[kk];
    if (act == 1 && acc < 0.f) acc = 0.f;   // relu
    Y[(size_t)b * N + n] = acc;
}

// =====================================================================
// Prep: swizzle f32 weights [N][srcStride] -> f16 WMMA B-fragments
// element j (0..15) of lane l holds K = kt*32 + (j>>3)*16 + (l>>4)*8 + (j&7), N = nt*16+(l&15)
// =====================================================================
__global__ void swizzle_w_kernel(const float* __restrict__ Wm, _Float16* __restrict__ dst,
                                 int K, int N, int colOff, int srcStride) {
    int i = blockIdx.x * blockDim.x + threadIdx.x;
    if (i >= K * N) return;
    int j    = i & 15;
    int lane = (i >> 4) & 31;
    int fidx = i >> 9;                // frag index = kt*(N/16) + nt
    int NT = N / 16;
    int nt = fidx % NT;
    int kt = fidx / NT;
    int k = kt * 32 + (j >> 3) * 16 + (lane >> 4) * 8 + (j & 7);
    int n = nt * 16 + (lane & 15);
    dst[i] = (_Float16)Wm[(size_t)n * srcStride + colOff + k];
}

// =====================================================================
// WMMA layer: Out[32 x N] = act(A[32 x K] * W + beff)
// 2 M-tiles per wave share each B fragment; 8 waves split N tiles.
// One-deep rotating prefetch of B fragments across the flattened tile loop.
// ACT: 0 = lipswish, 1 = tanh.  AF32: A read from f32 LDS (convert).
// =====================================================================
template <int K, int N, int ACT, bool AF32, bool OUTF32>
__device__ inline void wmma_layer(const void* __restrict__ sAv,
                                  const _Float16* __restrict__ frags,
                                  const float* __restrict__ sBeff,
                                  void* __restrict__ sOutv) {
    constexpr int KT  = K / 32;
    constexpr int NT  = N / 16;
    constexpr int NTW = NT / 8;       // N tiles per wave
    const int tid  = threadIdx.x;
    const int wave = tid >> 5;
    const int lane = tid & 31;
    const int ml = lane & 15;
    const int hh = lane >> 4;

    const _Float16* sAh = (const _Float16*)sAv;
    const float*    sAf = (const float*)sAv;
    _Float16* outH = (_Float16*)sOutv;
    float*    outF = (float*)sOutv;

    // A fragments for both M-tiles (ISA 16-bit 16x32 A layout).
    v16h aF0[KT], aF1[KT];
    #pragma unroll
    for (int kt = 0; kt < KT; ++kt) {
        #pragma unroll
        for (int i = 0; i < 8; ++i) {
            const int k0 = kt * 32 + hh * 8 + i;
            const int k1 = kt * 32 + 16 + hh * 8 + i;
            if constexpr (AF32) {
                aF0[kt][i]     = (_Float16)sAf[(size_t)ml * K + k0];
                aF0[kt][8 + i] = (_Float16)sAf[(size_t)ml * K + k1];
                aF1[kt][i]     = (_Float16)sAf[(size_t)(16 + ml) * K + k0];
                aF1[kt][8 + i] = (_Float16)sAf[(size_t)(16 + ml) * K + k1];
            } else {
                aF0[kt][i]     = sAh[(size_t)ml * K + k0];
                aF0[kt][8 + i] = sAh[(size_t)ml * K + k1];
                aF1[kt][i]     = sAh[(size_t)(16 + ml) * K + k0];
                aF1[kt][8 + i] = sAh[(size_t)(16 + ml) * K + k1];
            }
        }
    }

    // prefetch first B fragment (kt=0, nt=wave)
    v16h bF = load_bfrag(frags + ((size_t)wave * 32 + lane) * 16);

    #pragma unroll
    for (int q = 0; q < NTW; ++q) {
        const int nt = wave + q * 8;
        v8f c0 = {0.f, 0.f, 0.f, 0.f, 0.f, 0.f, 0.f, 0.f};
        v8f c1 = {0.f, 0.f, 0.f, 0.f, 0.f, 0.f, 0.f, 0.f};
        #pragma unroll
        for (int kt = 0; kt < KT; ++kt) {
            v16h bN = bF;
            if (kt + 1 < KT)
                bN = load_bfrag(frags + (((size_t)(kt + 1) * NT + nt) * 32 + lane) * 16);
            else if (q + 1 < NTW)
                bN = load_bfrag(frags + (((size_t)(nt + 8)) * 32 + lane) * 16);
            c0 = __builtin_amdgcn_wmma_f32_16x16x32_f16(false, aF0[kt], false, bF,
                                                        (short)0, c0, false, false);
            c1 = __builtin_amdgcn_wmma_f32_16x16x32_f16(false, aF1[kt], false, bF,
                                                        (short)0, c1, false, false);
            bF = bN;
        }
        const int n = nt * 16 + ml;
        const float be = sBeff[n];
        #pragma unroll
        for (int r = 0; r < 8; ++r) {
            const int m = r + 8 * hh;
            float v0 = c0[r] + be;
            float v1 = c1[r] + be;
            if constexpr (ACT == 0) { v0 = lipswish_f(v0); v1 = lipswish_f(v1); }
            else                    { v0 = tanh_f(v0);     v1 = tanh_f(v1); }
            if constexpr (OUTF32) {
                outF[(size_t)m * N + n]        = v0;
                outF[(size_t)(m + 16) * N + n] = v1;
            } else {
                outH[(size_t)m * N + n]        = (_Float16)v0;
                outH[(size_t)(m + 16) * N + n] = (_Float16)v1;
            }
        }
    }
}

// =====================================================================
// Main scan kernel: one WG = 32 batch rows, 8 wave32s
// =====================================================================
__global__ __launch_bounds__(256, 1) void cde_scan_kernel(
    const float* __restrict__ ts, const float* __restrict__ dX,
    const float* __restrict__ y0g,
    const _Float16* __restrict__ fVf1, const _Float16* __restrict__ fVf2,
    const _Float16* __restrict__ fVf3, const _Float16* __restrict__ fCf1,
    const _Float16* __restrict__ fCf2, const _Float16* __restrict__ fCf3,
    const float* __restrict__ vfW0raw, const float* __restrict__ vb0,
    const float* __restrict__ vb1, const float* __restrict__ vb2,
    const float* __restrict__ cfW0raw, const float* __restrict__ cb0,
    const float* __restrict__ cb1, const float* __restrict__ cb2,
    const float* __restrict__ roW, const float* __restrict__ roB,
    float* __restrict__ out) {

    __shared__ float    s_y[MR * HH];           // 16 KB
    __shared__ float    s_yhat[MR * HH];        // 16 KB (also A input for L1, f32)
    __shared__ float    s_f[MR * HH];           // 16 KB
    __shared__ _Float16 s_h1[MR * WW];          // 16 KB
    __shared__ _Float16 s_h2[MR * WW];          // 16 KB
    __shared__ _Float16 s_g[MR * HH * DD];      // 64 KB
    __shared__ float    s_dx[MR * DD];          // 1 KB
    __shared__ float    s_bev0[WW];             // step-dep eff bias, vf L1
    __shared__ float    s_bec0[WW];             // step-dep eff bias, cvf L1
    __shared__ float    s_bv1[WW], s_bv2[HH];
    __shared__ float    s_bc1[WW], s_bc2[HH * DD];
    __shared__ float    s_ro[HH];               // total ~154 KB

    const int tid = threadIdx.x;
    const int r0  = blockIdx.x * MR;

    // ---- stage static biases / readout ----
    s_bv1[tid] = vb1[tid];
    s_bc1[tid] = cb1[tid];
    if (tid < HH) { s_bv2[tid] = vb2[tid]; s_ro[tid] = roW[tid]; }
    for (int i = tid; i < HH * DD; i += 256) s_bc2[i] = cb2[i];

    // ---- load y0, init carry (y, yhat) ----
    for (int idx = tid; idx < MR * HH; idx += 256) {
        float v = y0g[(size_t)(r0 + (idx >> 7)) * HH + (idx & (HH - 1))];
        s_y[idx] = v; s_yhat[idx] = v;
    }
    float tcur = ts[(size_t)r0 * TT];
    const float rb = roB[0];

    // L1 effective biases at t0
    s_bev0[tid] = vb0[tid] + vfW0raw[(size_t)tid * (HH + 1)] * tcur;
    s_bec0[tid] = cb0[tid] + cfW0raw[(size_t)tid * (HH + 1)] * tcur;
    __syncthreads();

    // ---- f0 = vf(t0, y0) ; g0 = cvf(t0, y0) ----
    wmma_layer<HH,      WW, 0, true,  false>(s_yhat, fVf1, s_bev0, s_h1); __syncthreads();
    wmma_layer<WW,      WW, 0, false, false>(s_h1,   fVf2, s_bv1,  s_h2); __syncthreads();
    wmma_layer<WW,      HH, 1, false, true >(s_h2,   fVf3, s_bv2,  s_f);  __syncthreads();
    wmma_layer<HH,      WW, 0, true,  false>(s_yhat, fCf1, s_bec0, s_h1); __syncthreads();
    wmma_layer<WW,      WW, 0, false, false>(s_h1,   fCf2, s_bc1,  s_h2); __syncthreads();
    wmma_layer<WW, HH * DD, 1, false, false>(s_h2,   fCf3, s_bc2,  s_g);  __syncthreads();

    float upd[16], fold[16];

    // ---- scan over T steps ----
    for (int k = 0; k < TT; ++k) {
        const float t1 = tcur + DTF;

        // step-dependent staging: dX and L1 effective biases at t1
        s_dx[tid] = dX[((size_t)(r0 + (tid >> 3)) * TT + k) * DD + (tid & 7)];
        s_bev0[tid] = vb0[tid] + vfW0raw[(size_t)tid * (HH + 1)] * t1;
        s_bec0[tid] = cb0[tid] + cfW0raw[(size_t)tid * (HH + 1)] * t1;
        __syncthreads();

        // upd = g_c @ dX (contraction over D=8)
        #pragma unroll
        for (int i = 0; i < 16; ++i) {
            const int idx = tid + i * 256;
            const int m = idx >> 7, h = idx & (HH - 1);
            float acc = 0.f;
            #pragma unroll
            for (int d = 0; d < DD; ++d)
                acc += (float)s_g[(size_t)m * HH * DD + h * DD + d] * s_dx[m * DD + d];
            upd[i] = acc;
        }
        // predictor: yhat1 = 2*y - yhat + f*dt + g@dX ; save old f
        #pragma unroll
        for (int i = 0; i < 16; ++i) {
            const int idx = tid + i * 256;
            fold[i] = s_f[idx];
            s_yhat[idx] = 2.f * s_y[idx] - s_yhat[idx] + fold[i] * DTF + upd[i];
        }
        __syncthreads();

        // f1 = vf(t1, yhat1) -> s_f
        wmma_layer<HH, WW, 0, true,  false>(s_yhat, fVf1, s_bev0, s_h1); __syncthreads();
        wmma_layer<WW, WW, 0, false, false>(s_h1,   fVf2, s_bv1,  s_h2); __syncthreads();
        wmma_layer<WW, HH, 1, false, true >(s_h2,   fVf3, s_bv2,  s_f);  __syncthreads();

        // upd += (f + f1)*dt
        #pragma unroll
        for (int i = 0; i < 16; ++i) {
            const int idx = tid + i * 256;
            upd[i] += (fold[i] + s_f[idx]) * DTF;
        }

        // g1 = cvf(t1, yhat1) -> s_g
        wmma_layer<HH, WW, 0, true,  false>(s_yhat, fCf1, s_bec0, s_h1); __syncthreads();
        wmma_layer<WW, WW, 0, false, false>(s_h1,   fCf2, s_bc1,  s_h2); __syncthreads();
        wmma_layer<WW, HH * DD, 1, false, false>(s_h2, fCf3, s_bc2, s_g); __syncthreads();

        // corrector: y1 = y + 0.5*(upd + g1@dX)
        #pragma unroll
        for (int i = 0; i < 16; ++i) {
            const int idx = tid + i * 256;
            const int m = idx >> 7, h = idx & (HH - 1);
            float acc = 0.f;
            #pragma unroll
            for (int d = 0; d < DD; ++d)
                acc += (float)s_g[(size_t)m * HH * DD + h * DD + d] * s_dx[m * DD + d];
            s_y[idx] += 0.5f * (upd[i] + acc);
        }
        __syncthreads();

        // readout: out[b, k] = y1 . roW + rb
        if (tid < MR) {
            float acc = rb;
            #pragma unroll 4
            for (int h = 0; h < HH; ++h) acc += s_y[tid * HH + h] * s_ro[h];
            out[(size_t)(r0 + tid) * TT + k] = acc;
        }
        tcur = t1;
        __syncthreads();
    }
}

// =====================================================================
// launch
// =====================================================================
extern "C" void kernel_launch(void* const* d_in, const int* in_sizes, int n_in,
                              void* d_out, int out_size, void* d_ws, size_t ws_size,
                              hipStream_t stream) {
    (void)in_sizes; (void)n_in; (void)out_size; (void)ws_size;

    const float* ts   = (const float*)d_in[0];
    const float* ys   = (const float*)d_in[1];
    const float* iW0  = (const float*)d_in[2];   // [256][9]
    const float* iW1  = (const float*)d_in[3];   // [256][256]
    const float* iW2  = (const float*)d_in[4];   // [128][256]
    const float* iB0  = (const float*)d_in[5];
    const float* iB1  = (const float*)d_in[6];
    const float* iB2  = (const float*)d_in[7];
    const float* vW0  = (const float*)d_in[8];   // [256][129]
    const float* vW1  = (const float*)d_in[9];   // [256][256]
    const float* vW2  = (const float*)d_in[10];  // [128][256]
    const float* vB0  = (const float*)d_in[11];
    const float* vB1  = (const float*)d_in[12];
    const float* vB2  = (const float*)d_in[13];
    const float* cW0  = (const float*)d_in[14];  // [256][129]
    const float* cW1  = (const float*)d_in[15];  // [256][256]
    const float* cW2  = (const float*)d_in[16];  // [1024][256]
    const float* cB0  = (const float*)d_in[17];
    const float* cB1  = (const float*)d_in[18];
    const float* cB2  = (const float*)d_in[19];
    const float* roW  = (const float*)d_in[20];
    const float* roB  = (const float*)d_in[21];
    float* out = (float*)d_out;

    // ---- workspace carve ----
    char* ws = (char*)d_ws;
    size_t off = 0;
    float* dXbuf = (float*)(ws + off); off += (size_t)BB * TT * DD * 4;   // 8 MB
    float* y0buf = (float*)(ws + off); off += (size_t)BB * HH * 4;
    float* tmpA  = (float*)(ws + off); off += (size_t)BB * WW * 4;
    float* tmpB  = (float*)(ws + off); off += (size_t)BB * WW * 4;
    _Float16* fVf1 = (_Float16*)(ws + off); off += (size_t)HH * WW * 2;
    _Float16* fVf2 = (_Float16*)(ws + off); off += (size_t)WW * WW * 2;
    _Float16* fVf3 = (_Float16*)(ws + off); off += (size_t)WW * HH * 2;
    _Float16* fCf1 = (_Float16*)(ws + off); off += (size_t)HH * WW * 2;
    _Float16* fCf2 = (_Float16*)(ws + off); off += (size_t)WW * WW * 2;
    _Float16* fCf3 = (_Float16*)(ws + off); off += (size_t)WW * HH * DD * 2;

    // ---- dX precompute ----
    dx_kernel<<<(BB + 63) / 64, 64, 0, stream>>>(ts, ys, dXbuf);

    // ---- init MLP (f32) ----
    pack_init_kernel<<<(BB * 9 + 255) / 256, 256, 0, stream>>>(ts, ys, tmpB);
    dense_f32_kernel<<<(BB * WW + 255) / 256, 256, 0, stream>>>(tmpB, 9,  iW0, iB0, tmpA, 9,  WW, 1);
    dense_f32_kernel<<<(BB * WW + 255) / 256, 256, 0, stream>>>(tmpA, WW, iW1, iB1, tmpB, WW, WW, 1);
    dense_f32_kernel<<<(BB * HH + 255) / 256, 256, 0, stream>>>(tmpB, WW, iW2, iB2, y0buf, WW, HH, 0);

    // ---- weight swizzle to WMMA f16 B-fragments (t-column folded into bias) ----
    swizzle_w_kernel<<<(HH * WW      + 255) / 256, 256, 0, stream>>>(vW0, fVf1, HH, WW,      1, HH + 1);
    swizzle_w_kernel<<<(WW * WW      + 255) / 256, 256, 0, stream>>>(vW1, fVf2, WW, WW,      0, WW);
    swizzle_w_kernel<<<(WW * HH      + 255) / 256, 256, 0, stream>>>(vW2, fVf3, WW, HH,      0, WW);
    swizzle_w_kernel<<<(HH * WW      + 255) / 256, 256, 0, stream>>>(cW0, fCf1, HH, WW,      1, HH + 1);
    swizzle_w_kernel<<<(WW * WW      + 255) / 256, 256, 0, stream>>>(cW1, fCf2, WW, WW,      0, WW);
    swizzle_w_kernel<<<(WW * HH * DD + 255) / 256, 256, 0, stream>>>(cW2, fCf3, WW, HH * DD, 0, WW);

    // ---- main scan: 16 WGs x 8 wave32s, 32 batch rows each ----
    cde_scan_kernel<<<BB / MR, 256, 0, stream>>>(
        ts, dXbuf, y0buf,
        fVf1, fVf2, fVf3, fCf1, fCf2, fCf3,
        vW0, vB0, vB1, vB2,
        cW0, cB0, cB1, cB2,
        roW, roB, out);
}
